// AtlasTransformerV5b_64355789963647
// MI455X (gfx1250) — compile-verified, matrix operand order checked by reading, not run on hardware
//
#include <hip/hip_runtime.h>

// ============================================================================
// AtlasTransformer (graph transformer) for MI455X / gfx1250, wave32 + WMMA.
//
// Heavy GEMMs: bf16 inputs, fp32 accum via v_wmma_f32_16x16x32_bf16.
// GEMM K-loop uses a double-buffered LDS pipeline fed by CDNA5
// GLOBAL_LOAD_ASYNC_TO_LDS_B128 (ASYNCcnt), one barrier per K-step, with the
// next tile's DMA overlapped with the current tile's WMMA work.
// Weights pre-packed to bf16 transposed [N][K] so A and B^T tiles both load
// contiguously. Attention heads batched on blockIdx.z. Interaction head is
// fully fused (never materializes the 537MB (G,G,H) pair tensor).
//
// Input order assumes jax pytree (sorted-dict-key) flattening:
//  0 block_ids(512 i32) 1 edge(512*512*16) 2 node_features(512*32)
//  3 inter.W1(1040*512) 4 inter.W2(512*1) 5 inter.b1(512) 6 inter.b2(1)
//  7+20*l per layer, keys sorted:
//    +0 We1 +1 We2 +2 Wf1 +3 Wf2 +4 Wk +5 Wo +6 Wq +7 Wv
//    +8 be1 +9 be2 +10 bf1 +11 bf2 +12 bk +13 bo +14 bq +15 bv
//    +16 n1b +17 n1g +18 n2b +19 n2g
//  47 node.W(32*512) 48 node.b 49 node.beta 50 node.g
// Output: x(512*512) | interaction_scores(512*512) | attn_w(512*512), fp32.
// ============================================================================

typedef __bf16 bf16;
typedef __attribute__((ext_vector_type(8)))  bf16  v8bf;
typedef __attribute__((ext_vector_type(16))) bf16  v16bf;
typedef __attribute__((ext_vector_type(8)))  float v8f;

constexpr int G = 512, NODEF = 32, H = 512, HEADS = 8, EF = 16, HD = 64;

__device__ __forceinline__ float eluf(float x) {
  return x > 0.0f ? x : __expf(x) - 1.0f;
}

__device__ __forceinline__ v16bf cat16(v8bf lo, v8bf hi) {
  return __builtin_shufflevector(lo, hi, 0, 1, 2, 3, 4, 5, 6, 7,
                                 8, 9, 10, 11, 12, 13, 14, 15);
}

#define WMMA_BF16(a, b, c)                                                    \
  __builtin_amdgcn_wmma_f32_16x16x32_bf16(false, (a), false, (b), (short)0,   \
                                          (c), false, false)

// CDNA5 async memory->LDS DMA (ASYNCcnt tracked). LDS byte offset is the low
// 32 bits of the generic pointer (ISA 10.2: flat LDS addr = addr[31:0]).
// Copies 32B (two b128, INST_OFFSET applies to both LDS and global address).
__device__ __forceinline__ void async_cp32(const bf16* g, bf16* l) {
  unsigned int la = (unsigned int)(unsigned long long)l;
  unsigned long long ga = (unsigned long long)g;
  asm volatile("global_load_async_to_lds_b128 %0, %1, off"
               :: "v"(la), "v"(ga) : "memory");
  asm volatile("global_load_async_to_lds_b128 %0, %1, off offset:16"
               :: "v"(la), "v"(ga) : "memory");
}

__device__ __forceinline__ void wait_async0() {
  asm volatile("s_wait_asynccnt 0" ::: "memory");
}

// ---------------------------------------------------------------------------
// fp32 -> bf16 elementwise pack
__global__ void kpack(const float* __restrict__ s, bf16* __restrict__ d, int n) {
  int i = blockIdx.x * 256 + threadIdx.x;
  if (i < n) d[i] = (bf16)s[i];
}

// fp32 W[K][N] -> bf16 BT[N][KP], zero-padded K..KP
__global__ void kpackT(const float* __restrict__ W, bf16* __restrict__ BT,
                       int K, int N, int KP) {
  int i = blockIdx.x * 256 + threadIdx.x;
  if (i >= N * KP) return;
  int n = i / KP, k = i % KP;
  BT[i] = (k < K) ? (bf16)W[(size_t)k * N + n] : (bf16)0.0f;
}

__global__ void kcopy(const float* __restrict__ s, float* __restrict__ d, int n) {
  int i = blockIdx.x * 256 + threadIdx.x;
  if (i < n) d[i] = s[i];
}

// v(G,H) fp32 -> vT(HEADS,HD,G) bf16  (flat: vT[c*G+j] = v[j*H+c])
__global__ void kvt(const float* __restrict__ v, bf16* __restrict__ vT) {
  int idx = blockIdx.x * 256 + threadIdx.x;
  if (idx >= G * H) return;
  int c = idx / G, j = idx % G;
  vT[idx] = (bf16)v[(size_t)j * H + c];
}

// attn_w = mean over heads of fp32 attn probs
__global__ void kattnw(const float* __restrict__ A, float* __restrict__ out) {
  int idx = blockIdx.x * 256 + threadIdx.x;
  if (idx >= G * G) return;
  float s = 0.0f;
  #pragma unroll
  for (int h = 0; h < HEADS; h++) s += A[(size_t)h * G * G + idx];
  out[idx] = s * (1.0f / (float)HEADS);
}

// ---------------------------------------------------------------------------
// LayerNorm over H=512, optional ELU; writes bf16 (always) + fp32 (optional).
__global__ __launch_bounds__(256) void kln(const float* __restrict__ X,
                                           const float* __restrict__ g,
                                           const float* __restrict__ b,
                                           float* __restrict__ outF,
                                           bf16* __restrict__ outB, int doElu) {
  __shared__ float red[256];
  const int row = blockIdx.x, tid = threadIdx.x;
  const float* xr = X + (size_t)row * H;
  float x0 = xr[tid], x1 = xr[tid + 256];
  red[tid] = x0 + x1;
  __syncthreads();
  for (int s = 128; s > 0; s >>= 1) {
    if (tid < s) red[tid] += red[tid + s];
    __syncthreads();
  }
  float mean = red[0] * (1.0f / (float)H);
  __syncthreads();
  float d0 = x0 - mean, d1 = x1 - mean;
  red[tid] = d0 * d0 + d1 * d1;
  __syncthreads();
  for (int s = 128; s > 0; s >>= 1) {
    if (tid < s) red[tid] += red[tid + s];
    __syncthreads();
  }
  float inv = rsqrtf(red[0] * (1.0f / (float)H) + 1e-5f);
  float y0 = d0 * inv * g[tid] + b[tid];
  float y1 = d1 * inv * g[tid + 256] + b[tid + 256];
  if (doElu) { y0 = eluf(y0); y1 = eluf(y1); }
  outB[(size_t)row * H + tid] = (bf16)y0;
  outB[(size_t)row * H + tid + 256] = (bf16)y1;
  if (outF) {
    outF[(size_t)row * H + tid] = y0;
    outF[(size_t)row * H + tid + 256] = y1;
  }
}

// ---------------------------------------------------------------------------
// Edge bias MLP: eb[h][i][j] = (elu(e@We1+be1)@We2+be2)[h]. K=16: VALU.
__global__ void keb(const float* __restrict__ edge, const float* __restrict__ We1,
                    const float* __restrict__ be1, const float* __restrict__ We2,
                    const float* __restrict__ be2, float* __restrict__ eb) {
  int pair = blockIdx.x * 256 + threadIdx.x;
  if (pair >= G * G) return;
  const float* e = edge + (size_t)pair * EF;
  float ef[EF];
  #pragma unroll
  for (int i = 0; i < EF; i++) ef[i] = e[i];
  float hid[16];
  #pragma unroll
  for (int h2 = 0; h2 < 16; h2++) {
    float s = be1[h2];
    #pragma unroll
    for (int i = 0; i < EF; i++) s += ef[i] * We1[i * 16 + h2];
    hid[h2] = eluf(s);
  }
  #pragma unroll
  for (int hh = 0; hh < HEADS; hh++) {
    float s = be2[hh];
    #pragma unroll
    for (int h2 = 0; h2 < 16; h2++) s += hid[h2] * We2[h2 * HEADS + hh];
    eb[(size_t)hh * G * G + pair] = s;
  }
}

// ---------------------------------------------------------------------------
// Row softmax over G=512; in-place fp32 + bf16 probs copy.
__global__ __launch_bounds__(256) void ksoftmax(float* __restrict__ S,
                                                bf16* __restrict__ P) {
  __shared__ float red[256];
  const int i = blockIdx.x, h = blockIdx.y, tid = threadIdx.x;
  float* row = S + ((size_t)h * G + i) * G;
  float s0 = row[tid], s1 = row[tid + 256];
  red[tid] = fmaxf(s0, s1);
  __syncthreads();
  for (int s = 128; s > 0; s >>= 1) {
    if (tid < s) red[tid] = fmaxf(red[tid], red[tid + s]);
    __syncthreads();
  }
  float mx = red[0];
  __syncthreads();
  float e0 = __expf(s0 - mx), e1 = __expf(s1 - mx);
  red[tid] = e0 + e1;
  __syncthreads();
  for (int s = 128; s > 0; s >>= 1) {
    if (tid < s) red[tid] += red[tid + s];
    __syncthreads();
  }
  float inv = 1.0f / red[0];
  float p0 = e0 * inv, p1 = e1 * inv;
  row[tid] = p0;
  row[tid + 256] = p1;
  bf16* pr = P + ((size_t)h * G + i) * G;
  pr[tid] = (bf16)p0;
  pr[tid + 256] = (bf16)p1;
}

// ---------------------------------------------------------------------------
// Generic batched bf16 WMMA GEMM: C[z] = act(A[z] @ BT[z]^T + bias [+res]).
// 128 threads = 4 waves; 64x64 block tile, 32x32 per wave, K step 32.
// Double-buffered LDS fed by async memory->LDS DMA; one barrier per K-step.
// mode 0: standard epilogue.  mode 1: attention scores (scale + eb + mask).
__global__ __launch_bounds__(128) void kgemm(
    const bf16* __restrict__ A, int lda, long long sAz,
    const bf16* __restrict__ BT, int ldb, long long sBz,
    float* __restrict__ Cf, bf16* __restrict__ Cb, int ldc, long long sCz,
    const float* __restrict__ bias, const float* __restrict__ res,
    int M, int N, int K, int act, int mode,
    const float* __restrict__ ebp, long long sEz,
    const int* __restrict__ bids, float scale) {
  __shared__ bf16 lA[2][64][40];  // 80B row stride keeps 16B alignment
  __shared__ bf16 lB[2][64][40];
  const int z = blockIdx.z;
  A += (long long)z * sAz;
  BT += (long long)z * sBz;
  const int rowBase = blockIdx.y * 64;
  const int colBase = blockIdx.x * 64;
  const int tid = threadIdx.x;
  const int lane = tid & 31;
  const int wid = tid >> 5;
  const int wm = wid >> 1, wn = wid & 1;
  const int lr = lane & 15;
  const int kb = (lane < 16) ? 0 : 8;  // K-run base per ISA 16-bit A/B layout
  const int r = tid >> 1;
  const int c0 = (tid & 1) * 16;

  // Each thread DMAs 32B of A and 32B of B per tile (64x32 bf16 tiles).
  auto issueTile = [&](int k0, int buf) {
    async_cp32(A + (long long)(rowBase + r) * lda + k0 + c0, &lA[buf][r][c0]);
    async_cp32(BT + (long long)(colBase + r) * ldb + k0 + c0, &lB[buf][r][c0]);
  };

  v8f acc[2][2] = {};

  issueTile(0, 0);
  int buf = 0;
  for (int k0 = 0; k0 < K; k0 += 32) {
    wait_async0();       // our buffer's DMA landed in LDS
    __syncthreads();     // everyone's landed; prior reads of buf^1 done
    if (k0 + 32 < K) issueTile(k0 + 32, buf ^ 1);  // overlap next tile DMA

    v16bf af[2], bfr[2];
    #pragma unroll
    for (int m = 0; m < 2; m++) {
      int rr = wm * 32 + m * 16 + lr;
      af[m] = cat16(*(const v8bf*)&lA[buf][rr][kb],
                    *(const v8bf*)&lA[buf][rr][kb + 16]);
    }
    #pragma unroll
    for (int n = 0; n < 2; n++) {
      int rr = wn * 32 + n * 16 + lr;
      bfr[n] = cat16(*(const v8bf*)&lB[buf][rr][kb],
                     *(const v8bf*)&lB[buf][rr][kb + 16]);
    }
    #pragma unroll
    for (int m = 0; m < 2; m++)
      #pragma unroll
      for (int n = 0; n < 2; n++) acc[m][n] = WMMA_BF16(af[m], bfr[n], acc[m][n]);
    buf ^= 1;
  }

  const int rofs = (lane >> 4) << 3;  // C/D: lanes 16-31 hold rows M=v+8
  #pragma unroll
  for (int m = 0; m < 2; m++) {
    #pragma unroll
    for (int n = 0; n < 2; n++) {
      int gcol = colBase + wn * 32 + n * 16 + lr;
      #pragma unroll
      for (int v = 0; v < 8; v++) {
        int grow = rowBase + wm * 32 + m * 16 + v + rofs;
        long long idx = (long long)grow * ldc + gcol;
        float val = acc[m][n][v];
        if (mode == 1) {
          val = val * scale + ebp[(long long)z * sEz + (long long)grow * ldc + gcol];
          if (bids[grow] != bids[gcol]) val = -__builtin_inff();
          Cf[(long long)z * sCz + idx] = val;
        } else {
          if (bias) val += bias[gcol];
          if (res) val += res[idx];
          if (act) val = eluf(val);
          if (Cf) Cf[(long long)z * sCz + idx] = val;
          if (Cb) Cb[(long long)z * sCz + idx] = (bf16)val;
        }
      }
    }
  }
}

// ---------------------------------------------------------------------------
// Fused interaction head. Block = 16x16 pair tile (i-tile x j-tile), 4 waves,
// wave w owns i rows i0+4w..i0+4w+3. Per 16-wide h-chunk: WMMA eW tile
// (M=16 pairs, N=16 h, K=16 edge feats zero-padded to 32), add xi+xj+b1,
// ELU, multiply by W2[h], accumulate; lane-reduce at the end.
__global__ __launch_bounds__(128) void kinter(
    const bf16* __restrict__ edge_bf, const bf16* __restrict__ WeT,
    const float* __restrict__ xi, const float* __restrict__ xj,
    const float* __restrict__ b1, const float* __restrict__ W2,
    const float* __restrict__ b2, float* __restrict__ out) {
  const int j0 = blockIdx.x * 16;
  const int i0 = blockIdx.y * 16;
  const int lane = threadIdx.x & 31;
  const int wid = threadIdx.x >> 5;
  const int lr = lane & 15;
  const int kb = (lane < 16) ? 0 : 8;

  v8bf zero8 = {};
  v16bf af[4];  // loop-invariant A fragments: edge feats of 16 pairs each
  #pragma unroll
  for (int t = 0; t < 4; t++) {
    int i = i0 + wid * 4 + t;
    const bf16* ap = edge_bf + ((size_t)i * G + j0 + lr) * EF + kb;
    af[t] = cat16(*(const v8bf*)ap, zero8);  // K=16..31 are zero
  }

  float S[4][8] = {};
  for (int h0 = 0; h0 < H; h0 += 16) {
    int hcol = h0 + lr;
    v16bf bfr = cat16(*(const v8bf*)(WeT + (size_t)hcol * 32 + kb),
                      *(const v8bf*)(WeT + (size_t)hcol * 32 + kb + 16));
    float b1h = b1[hcol];
    float w2h = W2[hcol];
    #pragma unroll
    for (int t = 0; t < 4; t++) {
      v8f acc = {};
      acc = WMMA_BF16(af[t], bfr, acc);
      int i = i0 + wid * 4 + t;
      float xih = xi[(size_t)i * H + hcol];
      #pragma unroll
      for (int v = 0; v < 8; v++) {
        int m = v + ((lane >> 4) << 3);
        int j = j0 + m;
        float val = acc[v] + xih + xj[(size_t)j * H + hcol] + b1h;
        S[t][v] += eluf(val) * w2h;
      }
    }
  }

  // reduce across the 16 lanes of each half (h-columns)
  #pragma unroll
  for (int t = 0; t < 4; t++)
    #pragma unroll
    for (int v = 0; v < 8; v++) {
      float s = S[t][v];
      for (int off = 1; off < 16; off <<= 1) s += __shfl_xor(s, off, 32);
      S[t][v] = s;
    }
  if ((lane & 15) == 0) {
    int mofs = (lane >> 4) * 8;
    float bb = b2[0];
    #pragma unroll
    for (int t = 0; t < 4; t++) {
      int i = i0 + wid * 4 + t;
      #pragma unroll
      for (int v = 0; v < 8; v++)
        out[(size_t)i * G + j0 + v + mofs] = S[t][v] + bb;
    }
  }
}

// ===========================================================================
extern "C" void kernel_launch(void* const* d_in, const int* in_sizes, int n_in,
                              void* d_out, int out_size, void* d_ws,
                              size_t ws_size, hipStream_t stream) {
  (void)in_sizes; (void)n_in; (void)out_size; (void)ws_size;
  // ---- inputs (jax pytree sorted-key order) ----
  const int* block_ids = (const int*)d_in[0];
  const float* edge = (const float*)d_in[1];
  const float* nodef = (const float*)d_in[2];
  const float* i_W1 = (const float*)d_in[3];
  const float* i_W2 = (const float*)d_in[4];
  const float* i_b1 = (const float*)d_in[5];
  const float* i_b2 = (const float*)d_in[6];
  const float* nW = (const float*)d_in[47];
  const float* nb = (const float*)d_in[48];
  const float* nbeta = (const float*)d_in[49];
  const float* ng = (const float*)d_in[50];

  float* outX = (float*)d_out;
  float* outInter = outX + G * H;
  float* outAttnW = outX + G * H + G * G;

  // ---- workspace bump allocator ----
  char* wsp = (char*)d_ws;
  auto bump = [&](size_t bytes) -> void* {
    void* p = (void*)wsp;
    wsp += (bytes + 255) & ~(size_t)255;
    return p;
  };
  bf16* WnT = (bf16*)bump((size_t)H * NODEF * 2);
  bf16 *WqT[2], *WkT[2], *WvT[2], *WoT[2], *Wf1T[2], *Wf2T[2];
  for (int l = 0; l < 2; l++) {
    WqT[l] = (bf16*)bump((size_t)H * H * 2);
    WkT[l] = (bf16*)bump((size_t)H * H * 2);
    WvT[l] = (bf16*)bump((size_t)H * H * 2);
    WoT[l] = (bf16*)bump((size_t)H * H * 2);
    Wf1T[l] = (bf16*)bump((size_t)2 * H * H * 2);
    Wf2T[l] = (bf16*)bump((size_t)H * 2 * H * 2);
  }
  bf16* WiT = (bf16*)bump((size_t)H * H * 2);
  bf16* WjT = (bf16*)bump((size_t)H * H * 2);
  bf16* WeTp = (bf16*)bump((size_t)H * 32 * 2);  // K=16 padded to 32
  bf16* node_bf = (bf16*)bump((size_t)G * NODEF * 2);
  bf16* edge_bf = (bf16*)bump((size_t)G * G * EF * 2);
  float* x_f = (float*)bump((size_t)G * H * 4);
  bf16* x_bf = (bf16*)bump((size_t)G * H * 2);
  bf16* n_bf = (bf16*)bump((size_t)G * H * 2);
  bf16* q_bf = (bf16*)bump((size_t)G * H * 2);
  bf16* k_bf = (bf16*)bump((size_t)G * H * 2);
  float* v_f = (float*)bump((size_t)G * H * 4);
  bf16* vT_bf = (bf16*)bump((size_t)G * H * 2);
  bf16* out_bf = (bf16*)bump((size_t)G * H * 2);
  bf16* f1_bf = (bf16*)bump((size_t)G * 2 * H * 2);
  float* tmp_f = (float*)bump((size_t)G * H * 4);
  float* scores = (float*)bump((size_t)HEADS * G * G * 4);
  bf16* attn_bf = (bf16*)bump((size_t)HEADS * G * G * 2);
  float* eb_f = (float*)bump((size_t)HEADS * G * G * 4);
  float* xi_f = (float*)bump((size_t)G * H * 4);
  float* xj_f = (float*)bump((size_t)G * H * 4);

  auto gs = [](long long n) { return (unsigned)((n + 255) / 256); };
  const long long GG = (long long)G * G;

  // ---- weight packing (bf16, transposed to [N][K]) ----
  kpackT<<<gs((long long)H * NODEF), 256, 0, stream>>>(nW, WnT, NODEF, H, NODEF);
  for (int l = 0; l < 2; l++) {
    int bidx = 7 + 20 * l;
    const float* Wq = (const float*)d_in[bidx + 6];
    const float* Wk = (const float*)d_in[bidx + 4];
    const float* Wv = (const float*)d_in[bidx + 7];
    const float* Wo = (const float*)d_in[bidx + 5];
    const float* Wf1 = (const float*)d_in[bidx + 2];
    const float* Wf2 = (const float*)d_in[bidx + 3];
    kpackT<<<gs((long long)H * H), 256, 0, stream>>>(Wq, WqT[l], H, H, H);
    kpackT<<<gs((long long)H * H), 256, 0, stream>>>(Wk, WkT[l], H, H, H);
    kpackT<<<gs((long long)H * H), 256, 0, stream>>>(Wv, WvT[l], H, H, H);
    kpackT<<<gs((long long)H * H), 256, 0, stream>>>(Wo, WoT[l], H, H, H);
    kpackT<<<gs((long long)2 * H * H), 256, 0, stream>>>(Wf1, Wf1T[l], H, 2 * H, H);
    kpackT<<<gs((long long)2 * H * H), 256, 0, stream>>>(Wf2, Wf2T[l], 2 * H, H, 2 * H);
  }
  kpackT<<<gs((long long)H * H), 256, 0, stream>>>(i_W1, WiT, H, H, H);
  kpackT<<<gs((long long)H * H), 256, 0, stream>>>(i_W1 + (size_t)H * H, WjT, H, H, H);
  kpackT<<<gs((long long)H * 32), 256, 0, stream>>>(i_W1 + (size_t)2 * H * H, WeTp, EF, H, 32);
  kpack<<<gs((long long)G * NODEF), 256, 0, stream>>>(nodef, node_bf, G * NODEF);
  kpack<<<gs(GG * EF), 256, 0, stream>>>(edge, edge_bf, G * G * EF);

  // ---- embedding: x = elu(LN(node @ Wn + bn)) ----
  kgemm<<<dim3(H / 64, G / 64, 1), 128, 0, stream>>>(
      node_bf, NODEF, 0, WnT, NODEF, 0, tmp_f, nullptr, H, 0, nb, nullptr,
      G, H, NODEF, 0, 0, nullptr, 0, nullptr, 0.0f);
  kln<<<G, 256, 0, stream>>>(tmp_f, ng, nbeta, x_f, x_bf, 1);

  // ---- transformer layers ----
  for (int l = 0; l < 2; l++) {
    int bidx = 7 + 20 * l;
    const float* We1 = (const float*)d_in[bidx + 0];
    const float* We2 = (const float*)d_in[bidx + 1];
    const float* be1 = (const float*)d_in[bidx + 8];
    const float* be2 = (const float*)d_in[bidx + 9];
    const float* bf1 = (const float*)d_in[bidx + 10];
    const float* bf2 = (const float*)d_in[bidx + 11];
    const float* bk = (const float*)d_in[bidx + 12];
    const float* bo = (const float*)d_in[bidx + 13];
    const float* bq = (const float*)d_in[bidx + 14];
    const float* bv = (const float*)d_in[bidx + 15];
    const float* n1b = (const float*)d_in[bidx + 16];
    const float* n1g = (const float*)d_in[bidx + 17];
    const float* n2b = (const float*)d_in[bidx + 18];
    const float* n2g = (const float*)d_in[bidx + 19];

    kln<<<G, 256, 0, stream>>>(x_f, n1g, n1b, nullptr, n_bf, 0);
    kgemm<<<dim3(H / 64, G / 64, 1), 128, 0, stream>>>(
        n_bf, H, 0, WqT[l], H, 0, nullptr, q_bf, H, 0, bq, nullptr,
        G, H, H, 0, 0, nullptr, 0, nullptr, 0.0f);
    kgemm<<<dim3(H / 64, G / 64, 1), 128, 0, stream>>>(
        n_bf, H, 0, WkT[l], H, 0, nullptr, k_bf, H, 0, bk, nullptr,
        G, H, H, 0, 0, nullptr, 0, nullptr, 0.0f);
    kgemm<<<dim3(H / 64, G / 64, 1), 128, 0, stream>>>(
        n_bf, H, 0, WvT[l], H, 0, v_f, nullptr, H, 0, bv, nullptr,
        G, H, H, 0, 0, nullptr, 0, nullptr, 0.0f);
    kvt<<<gs((long long)G * H), 256, 0, stream>>>(v_f, vT_bf);
    keb<<<gs(GG), 256, 0, stream>>>(edge, We1, be1, We2, be2, eb_f);
    // scores[h] = q_h @ k_h^T / sqrt(HD) + eb[h], masked
    kgemm<<<dim3(G / 64, G / 64, HEADS), 128, 0, stream>>>(
        q_bf, H, HD, k_bf, H, HD, scores, nullptr, G, GG, nullptr, nullptr,
        G, G, HD, 0, 1, eb_f, GG, block_ids, 0.125f);
    ksoftmax<<<dim3(G, HEADS), 256, 0, stream>>>(scores, attn_bf);
    if (l == 1) kattnw<<<gs(GG), 256, 0, stream>>>(scores, outAttnW);
    // out[h] = attn_h @ v_h
    kgemm<<<dim3(1, G / 64, HEADS), 128, 0, stream>>>(
        attn_bf, G, GG, vT_bf, G, (long long)HD * G, nullptr, out_bf, H, HD,
        nullptr, nullptr, G, HD, G, 0, 0, nullptr, 0, nullptr, 0.0f);
    // x += out @ Wo + bo
    kgemm<<<dim3(H / 64, G / 64, 1), 128, 0, stream>>>(
        out_bf, H, 0, WoT[l], H, 0, x_f, x_bf, H, 0, bo, x_f,
        G, H, H, 0, 0, nullptr, 0, nullptr, 0.0f);
    // FFN
    kln<<<G, 256, 0, stream>>>(x_f, n2g, n2b, nullptr, n_bf, 0);
    kgemm<<<dim3(2 * H / 64, G / 64, 1), 128, 0, stream>>>(
        n_bf, H, 0, Wf1T[l], H, 0, nullptr, f1_bf, 2 * H, 0, bf1, nullptr,
        G, 2 * H, H, 1, 0, nullptr, 0, nullptr, 0.0f);
    kgemm<<<dim3(H / 64, G / 64, 1), 128, 0, stream>>>(
        f1_bf, 2 * H, 0, Wf2T[l], 2 * H, 0, x_f, x_bf, H, 0, bf2, x_f,
        G, H, 2 * H, 0, 0, nullptr, 0, nullptr, 0.0f);
  }

  // ---- outputs ----
  kcopy<<<gs((long long)G * H), 256, 0, stream>>>(x_f, outX, G * H);
  kgemm<<<dim3(H / 64, G / 64, 1), 128, 0, stream>>>(
      x_bf, H, 0, WiT, H, 0, xi_f, nullptr, H, 0, nullptr, nullptr,
      G, H, H, 0, 0, nullptr, 0, nullptr, 0.0f);
  kgemm<<<dim3(H / 64, G / 64, 1), 128, 0, stream>>>(
      x_bf, H, 0, WjT, H, 0, xj_f, nullptr, H, 0, nullptr, nullptr,
      G, H, H, 0, 0, nullptr, 0, nullptr, 0.0f);
  kinter<<<dim3(G / 16, G / 16), 128, 0, stream>>>(
      edge_bf, WeTp, xi_f, xj_f, i_b1, i_W2, i_b2, outInter);
}
